// SingleClassGCNModel_shapLike_2_21878563406402
// MI455X (gfx1250) — matrix operand order repaired and checked by reading, not compiled, not run on hardware
//
#include <hip/hip_runtime.h>
#include <hip/hip_bf16.h>
#include <stdint.h>

// ---------------------------------------------------------------------------
// GNN forward (PyG GraphConv max-aggr x7 + BN + mean-pool + MLP head) for
// MI455X / gfx1250.
//   * activations bf16 (<=25.6 MB -> L2-resident; halves gather traffic)
//   * edge max-aggregation: order-preserving u32 encoding + atomicMax_u32
//   * finalize pass converts encoded agg -> plain bf16 (-inf -> 0), so the
//     WMMA GEMM reads both A operands as raw bf16 (no per-element decode)
//   * node GEMMs: v_wmma_f32_16x16x32_bf16, fusing rel + root + bias
//     (+PReLU for non-BN layers); BN layers emit f32 pre-act -> reduce -> norm
// d_in index map (setup_inputs insertion order):
//   0:x  1..21: conv{1..7}.{Wrel,brel,Wroot}  22,23: bn3.{gamma,beta}
//   24,25: bn5.{gamma,beta}  26: prelu_a  27..34: linear{0..3}.{W,b}
//   35: edge_index[2,E]  36: batch[N]
// ---------------------------------------------------------------------------

typedef unsigned short u16;
typedef unsigned int   u32;
typedef __attribute__((ext_vector_type(16))) __bf16 v16bf;
typedef __attribute__((ext_vector_type(8)))  float  v8f;

#define N_NODES  50000
#define N_EDGES  800000
#define N_GRAPHS 256
#define EPS_BN   1e-5f
#define ENC_NEGINF 0x007FFFFFu   // enc(-inf), low bits only hit by init value

// order-preserving encode of a bf16 payload into u32 (monotone in value)
__device__ __forceinline__ u32 enc_bf(u16 bf) {
  return (bf & 0x8000u) ? ((u32)((u16)~bf) << 16)
                        : (((u32)bf | 0x8000u) << 16);
}
// inverse (values known to be bf16-derived; untouched init -> 0)
__device__ __forceinline__ u16 dec_bf(u32 e) {
  u16 t  = (u16)(e >> 16);
  u16 bf = (t & 0x8000u) ? (u16)(t ^ 0x8000u) : (u16)~t;
  return (e == ENC_NEGINF) ? (u16)0 : bf;
}
__device__ __forceinline__ u16 f2bf(float f) {    // round-to-nearest-even
  u32 u = __float_as_uint(f);
  u += 0x7FFFu + ((u >> 16) & 1u);
  return (u16)(u >> 16);
}
__device__ __forceinline__ float bf2f(u16 h) {
  return __uint_as_float(((u32)h) << 16);
}

// ------------------------------ utility kernels ----------------------------

__global__ void k_fill_u32(u32* __restrict__ p, u32 v, long long n) {
  long long i = blockIdx.x * (long long)blockDim.x + threadIdx.x;
  if (i < n) p[i] = v;
}

// W[dout][di] f32  ->  Bt[kpad][dout] bf16 (pre-transposed, K zero-padded)
__global__ void k_prep_w(const float* __restrict__ W, u16* __restrict__ Bt,
                         int di, int dout, int kpad) {
  int i = blockIdx.x * blockDim.x + threadIdx.x;
  if (i >= kpad * dout) return;
  int k = i / dout, n = i % dout;
  Bt[i] = f2bf((k < di) ? W[n * di + k] : 0.f);
}

// x[N,9] f32 -> h[N,32] bf16 zero-padded
__global__ void k_prep_x(const float* __restrict__ x, u16* __restrict__ h) {
  int i = blockIdx.x * blockDim.x + threadIdx.x;
  if (i >= N_NODES * 32) return;
  int n = i >> 5, k = i & 31;
  h[i] = f2bf((k < 9) ? x[n * 9 + k] : 0.f);
}

// ------------------------------ edge scatter-max ---------------------------
// one thread = one edge x 8 channels; one b128 bf16 read, 8x atomicMax_u32
__global__ void k_scatter_max(const u16* __restrict__ h, const int* __restrict__ ei,
                              u32* __restrict__ agg, int S) {
  long long i = blockIdx.x * (long long)blockDim.x + threadIdx.x;
  int cq = S >> 3;
  long long tot = (long long)N_EDGES * cq;
  if (i >= tot) return;
  int e = (int)(i / cq);
  int c = ((int)(i % cq)) << 3;
  int src = ei[e];
  int dst = ei[N_EDGES + e];
  uint4 q = *(const uint4*)(h + (size_t)src * S + c);
  u32 w[4] = {q.x, q.y, q.z, q.w};
  u32* ap = agg + (size_t)dst * S + c;
#pragma unroll
  for (int j = 0; j < 4; ++j) {
    atomicMax(&ap[2 * j],     enc_bf((u16)(w[j] & 0xFFFFu)));
    atomicMax(&ap[2 * j + 1], enc_bf((u16)(w[j] >> 16)));
  }
}

// encoded u32 agg -> packed bf16 agg (isolated-node -inf -> 0)
__global__ void k_agg_finalize(const u32* __restrict__ enc, u32* __restrict__ out,
                               long long npairs) {
  long long i = blockIdx.x * (long long)blockDim.x + threadIdx.x;
  if (i >= npairs) return;
  u32 e0 = enc[2 * i], e1 = enc[2 * i + 1];
  out[i] = (u32)dec_bf(e0) | ((u32)dec_bf(e1) << 16);
}

// ------------------------------ fused conv GEMM ----------------------------
// out[m][n] = sum_k agg[m][k]*Wrel[n][k] + sum_k h[m][k]*Wroot[n][k] + brel[n]
// 4 waves/block, each wave owns a 16x16 D tile; K in steps of 32 (bf16 WMMA).
union V16 { uint4 q[2]; u16 s[16]; v16bf v; };

template <bool WRITE_F32>
__global__ void k_conv_gemm(const u16* __restrict__ aggbf, const u16* __restrict__ hin,
                            const u16* __restrict__ Brel, const u16* __restrict__ Broot,
                            const float* __restrict__ bias, const float* __restrict__ pa,
                            u16* __restrict__ hout, float* __restrict__ f32out,
                            int Kpad, int Dout) {
  int lane = threadIdx.x & 31;
  int wave = threadIdx.x >> 5;
  int r0 = blockIdx.x << 4;                    // N_NODES % 16 == 0
  int n0 = ((blockIdx.y << 2) + wave) << 4;
  if (n0 >= Dout) return;                      // wave-uniform exit
  int mrow  = r0 + (lane & 15);                // A layout: both lane halves M=0..15
  int kbase = (lane >> 4) << 3;                // K chunks {k0..k0+7, k0+16..k0+23}
  const u16* ap = aggbf + (size_t)mrow * Kpad + kbase;
  const u16* hp = hin   + (size_t)mrow * Kpad + kbase;
  const u16* bp = Brel  + (size_t)kbase * Dout + n0 + (lane & 15);
  const u16* tp = Broot + (size_t)kbase * Dout + n0 + (lane & 15);
  size_t bstep = (size_t)32 * Dout;
  v8f acc = {0.f, 0.f, 0.f, 0.f, 0.f, 0.f, 0.f, 0.f};
  for (int kt = 0; kt < Kpad; kt += 32) {
    V16 aA, rA, bR, bT;
    aA.q[0] = *(const uint4*)ap;       aA.q[1] = *(const uint4*)(ap + 16);
    rA.q[0] = *(const uint4*)hp;       rA.q[1] = *(const uint4*)(hp + 16);
    for (int s = 0; s < 16; ++s) {
      size_t ko = (size_t)((s & 7) + ((s >> 3) << 4)) * Dout;
      bR.s[s] = bp[ko];                bT.s[s] = tp[ko];
    }
    acc = __builtin_amdgcn_wmma_f32_16x16x32_bf16(false, aA.v, false, bR.v,
                                                  (short)0, acc, false, false);
    acc = __builtin_amdgcn_wmma_f32_16x16x32_bf16(false, rA.v, false, bT.v,
                                                  (short)0, acc, false, false);
    ap += 32; hp += 32; bp += bstep; tp += bstep;
  }
  // D layout: lane -> n = lane&15 ; vgpr v -> m = v + 8*(lane>=16)
  int ncol  = n0 + (lane & 15);
  int mbase = r0 + ((lane >> 4) << 3);
  float b = bias[ncol];
  float a = pa[0];
  size_t idx = (size_t)mbase * Dout + ncol;
#pragma unroll
  for (int v = 0; v < 8; ++v) {
    float val = acc[v] + b;
    if (WRITE_F32) {
      f32out[idx] = val;                                 // BN layers: pre-act
    } else {
      val = (val >= 0.f) ? val : a * val;                // PReLU
      hout[idx] = f2bf(val);
    }
    idx += (size_t)Dout;
  }
}

// ------------------------------ batchnorm ----------------------------------
__global__ void k_bn_stats(const float* __restrict__ pre, float* __restrict__ st, int C) {
  int c    = threadIdx.x % C;
  int rpb  = blockDim.x / C;
  int r    = blockIdx.x * rpb + threadIdx.x / C;
  int step = gridDim.x * rpb;
  float s = 0.f, q = 0.f;
  for (; r < N_NODES; r += step) {
    float v = pre[(size_t)r * C + c];
    s += v; q += v * v;
  }
  atomicAdd(&st[c], s);
  atomicAdd(&st[C + c], q);
}

__global__ void k_bn_apply(const float* __restrict__ pre, const float* __restrict__ st,
                           const float* __restrict__ gamma, const float* __restrict__ beta,
                           const float* __restrict__ pa, u16* __restrict__ hout, int C) {
  long long i = blockIdx.x * (long long)blockDim.x + threadIdx.x;
  if (i >= (long long)N_NODES * C) return;
  int c = (int)(i % C);
  float inv_n = 1.f / (float)N_NODES;
  float mu  = st[c] * inv_n;
  float var = st[C + c] * inv_n - mu * mu;
  float v = (pre[i] - mu) * rsqrtf(var + EPS_BN) * gamma[c] + beta[c];
  float a = pa[0];
  v = (v >= 0.f) ? v : a * v;
  hout[i] = f2bf(v);
}

// ------------------------------ pooling + head -----------------------------
__global__ void k_pool(const u16* __restrict__ h, const int* __restrict__ batch,
                       float* __restrict__ pooled, float* __restrict__ cnt) {
  int i = blockIdx.x * blockDim.x + threadIdx.x;
  if (i >= N_NODES * 32) return;
  int n = i >> 5, c = i & 31;
  int g = batch[n];
  atomicAdd(&pooled[g * 32 + c], bf2f(h[i]));
  if (c == 0) atomicAdd(&cnt[g], 1.f);
}

__global__ void k_head(const float* __restrict__ pooled, const float* __restrict__ cnt,
                       const float* __restrict__ W0, const float* __restrict__ b0,
                       const float* __restrict__ W1, const float* __restrict__ b1,
                       const float* __restrict__ W2, const float* __restrict__ b2,
                       const float* __restrict__ W3, const float* __restrict__ b3,
                       const float* __restrict__ pa, float* __restrict__ out) {
  __shared__ float v0[32], v1[32], v2[16];
  int g = blockIdx.x, t = threadIdx.x;
  float a = pa[0];
  if (t < 32) v0[t] = pooled[g * 32 + t] / fmaxf(cnt[g], 1.f);
  __syncthreads();
  if (t < 32) {                                    // linear0 32->32 + PReLU
    float s = b0[t];
    for (int k = 0; k < 32; ++k) s += W0[t * 32 + k] * v0[k];
    v1[t] = (s >= 0.f) ? s : a * s;
  }
  __syncthreads();
  if (t < 32) {                                    // linear1 32->32 + PReLU
    float s = b1[t];
    for (int k = 0; k < 32; ++k) s += W1[t * 32 + k] * v1[k];
    v0[t] = (s >= 0.f) ? s : a * s;
  }
  __syncthreads();
  if (t < 16) {                                    // linear2 32->16 + PReLU
    float s = b2[t];
    for (int k = 0; k < 32; ++k) s += W2[t * 32 + k] * v0[k];
    v2[t] = (s >= 0.f) ? s : a * s;
  }
  __syncthreads();
  if (t == 0) {                                    // linear3 16->1, sigmoid
    float z = b3[0];
    for (int k = 0; k < 16; ++k) z += W3[k] * v2[k];
    out[g]            = 1.f / (1.f + __expf(-(z - 0.5f)));  // y
    out[N_GRAPHS + g] = z;                                  // logits
  }
}

// ------------------------------ host orchestration -------------------------
extern "C" void kernel_launch(void* const* d_in, const int* in_sizes, int n_in,
                              void* d_out, int out_size, void* d_ws, size_t ws_size,
                              hipStream_t stream) {
  (void)in_sizes; (void)n_in; (void)out_size; (void)ws_size;
  const float* x     = (const float*)d_in[0];
  const int*   ei    = (const int*)d_in[35];
  const int*   batch = (const int*)d_in[36];
  const float* pa    = (const float*)d_in[26];

  struct LCfg { int di, kpad, dout, wrel, brel, wroot, bn, gidx, bidx; };
  static const LCfg L[7] = {
    {  9,  32,  32,  1,  2,  3, 0,  0,  0},
    { 32,  32,  64,  4,  5,  6, 0,  0,  0},
    { 64,  64, 128,  7,  8,  9, 1, 22, 23},
    {128, 128, 256, 10, 11, 12, 0,  0,  0},
    {256, 256, 128, 13, 14, 15, 0,  0,  0},
    {128, 128,  64, 16, 17, 18, 0,  0,  0},
    { 64,  64,  32, 19, 20, 21, 1, 24, 25},
  };

  // bump-allocate workspace (total ~155 MB), 256B aligned
  char* ws = (char*)d_ws;
  size_t off = 0;
  auto alloc = [&](size_t bytes) -> char* {
    char* p = ws + off;
    off = (off + bytes + 255) & ~(size_t)255;
    return p;
  };
  u16*   hA     = (u16*)alloc((size_t)N_NODES * 256 * 2);
  u16*   hB     = (u16*)alloc((size_t)N_NODES * 256 * 2);
  u32*   agg    = (u32*)alloc((size_t)N_NODES * 256 * 4);   // encoded
  u16*   aggbf  = (u16*)alloc((size_t)N_NODES * 256 * 2);   // finalized bf16
  float* bnbuf  = (float*)alloc((size_t)N_NODES * 128 * 4);
  float* st     = (float*)alloc(2 * 128 * 4);
  float* pooled = (float*)alloc(N_GRAPHS * 32 * 4);
  float* cnt    = (float*)alloc(N_GRAPHS * 4);
  u16* wts[7][2];
  for (int l = 0; l < 7; ++l) {
    wts[l][0] = (u16*)alloc((size_t)L[l].kpad * L[l].dout * 2);
    wts[l][1] = (u16*)alloc((size_t)L[l].kpad * L[l].dout * 2);
  }

  // prep: weights -> transposed/padded bf16 ; x -> padded bf16
  for (int l = 0; l < 7; ++l) {
    int tot = L[l].kpad * L[l].dout;
    int blk = (tot + 255) / 256;
    k_prep_w<<<blk, 256, 0, stream>>>((const float*)d_in[L[l].wrel],  wts[l][0],
                                      L[l].di, L[l].dout, L[l].kpad);
    k_prep_w<<<blk, 256, 0, stream>>>((const float*)d_in[L[l].wroot], wts[l][1],
                                      L[l].di, L[l].dout, L[l].kpad);
  }
  k_prep_x<<<(N_NODES * 32 + 255) / 256, 256, 0, stream>>>(x, hA);

  u16* cur = hA;
  u16* nxt = hB;
  for (int l = 0; l < 7; ++l) {
    int S = L[l].kpad, D = L[l].dout;
    long long na = (long long)N_NODES * S;
    k_fill_u32<<<(unsigned)((na + 255) / 256), 256, 0, stream>>>(agg, ENC_NEGINF, na);
    long long ne = (long long)N_EDGES * (S >> 3);
    k_scatter_max<<<(unsigned)((ne + 255) / 256), 256, 0, stream>>>(cur, ei, agg, S);
    long long np = na >> 1;
    k_agg_finalize<<<(unsigned)((np + 255) / 256), 256, 0, stream>>>(agg, (u32*)aggbf, np);
    dim3 g((N_NODES + 15) / 16, (D + 63) / 64);
    if (L[l].bn) {
      k_conv_gemm<true><<<g, 128, 0, stream>>>(aggbf, cur, wts[l][0], wts[l][1],
                                               (const float*)d_in[L[l].brel], pa,
                                               nxt, bnbuf, S, D);
      k_fill_u32<<<1, 256, 0, stream>>>((u32*)st, 0u, 2 * D);
      k_bn_stats<<<256, 256, 0, stream>>>(bnbuf, st, D);
      long long tot = (long long)N_NODES * D;
      k_bn_apply<<<(unsigned)((tot + 255) / 256), 256, 0, stream>>>(
          bnbuf, st, (const float*)d_in[L[l].gidx], (const float*)d_in[L[l].bidx],
          pa, nxt, D);
    } else {
      k_conv_gemm<false><<<g, 128, 0, stream>>>(aggbf, cur, wts[l][0], wts[l][1],
                                                (const float*)d_in[L[l].brel], pa,
                                                nxt, bnbuf, S, D);
    }
    u16* t = cur; cur = nxt; nxt = t;
  }

  // mean pool + head
  k_fill_u32<<<(N_GRAPHS * 32 + 255) / 256, 256, 0, stream>>>((u32*)pooled, 0u, N_GRAPHS * 32);
  k_fill_u32<<<1, 256, 0, stream>>>((u32*)cnt, 0u, N_GRAPHS);
  k_pool<<<(N_NODES * 32 + 255) / 256, 256, 0, stream>>>(cur, batch, pooled, cnt);
  k_head<<<N_GRAPHS, 32, 0, stream>>>(pooled, cnt,
      (const float*)d_in[27], (const float*)d_in[28],
      (const float*)d_in[29], (const float*)d_in[30],
      (const float*)d_in[31], (const float*)d_in[32],
      (const float*)d_in[33], (const float*)d_in[34],
      pa, (float*)d_out);
}
